// GPTModel_26499948216516
// MI455X (gfx1250) — matrix operand order repaired and checked by reading, not compiled, not run on hardware
//
#include <hip/hip_runtime.h>
#include <hip/hip_bf16.h>
#include <math.h>

// ---- problem constants (match reference) ----
#define VOCAB  50257
#define DMODEL 768
#define SEQ    1024
#define NLAYER 4
#define NHEAD  12
#define BATCH  2
#define HDIM   64                 // DMODEL / NHEAD
#define NROW   (BATCH * SEQ)      // 2048 token rows
#define D3     (3 * DMODEL)       // 2304
#define D4     (4 * DMODEL)       // 3072

typedef __bf16 bf16;
typedef unsigned int u32;
typedef __attribute__((ext_vector_type(8)))  __bf16 v8bf;
typedef __attribute__((ext_vector_type(16))) __bf16 v16bf;
typedef __attribute__((ext_vector_type(8)))  float  v8f;
typedef __attribute__((ext_vector_type(4)))  u32    u32x4;
typedef __attribute__((ext_vector_type(4)))  int    i32x4;
typedef __attribute__((ext_vector_type(8)))  int    i32x8;

#define CEILDIV(a, b) (((a) + (b) - 1) / (b))

// =====================================================================
// Tensor Data Mover: DMA a [rows x tileK] bf16 tile (row stride = strideK
// elements) from global into LDS at byte offset ldsOff, padding each
// 32-DWORD row with 8 DWORDs (row pitch 160B) per ISA 8.3/8.4.
// =====================================================================
__device__ __forceinline__ void tdm_load_tile(u32 ldsOff, const bf16* gptr,
                                              u32 tileK, u32 rows, u32 strideK) {
  unsigned long long ga = (unsigned long long)(size_t)gptr;
  u32x4 g0;
  g0[0] = 1u;                                        // count=1 (valid, user mode)
  g0[1] = ldsOff;                                    // lds byte address
  g0[2] = (u32)(ga & 0xffffffffu);                   // global_addr[31:0]
  g0[3] = (u32)((ga >> 32) & 0x1ffffffu) | (2u << 30); // global_addr[56:32] | type=2
  i32x8 g1;
  g1[0] = (int)((1u << 16)      // data_size = 1 (2 bytes)
              | (1u << 20)      // pad_enable
              | (4u << 22)      // pad_interval: 32 DWORDs
              | (7u << 25));    // pad_amount: 8 DWORDs
  g1[1] = (int)((strideK & 0xffffu) << 16);                         // tensor_dim0 lo16
  g1[2] = (int)(((strideK >> 16) & 0xffffu) | ((rows & 0xffffu) << 16)); // dim0 hi | tensor_dim1 lo
  g1[3] = (int)(((rows >> 16) & 0xffffu) | ((tileK & 0xffffu) << 16));   // dim1 hi | tile_dim0
  g1[4] = (int)(rows & 0xffffu);                                    // tile_dim1 (tile_dim2=0)
  g1[5] = (int)strideK;                                             // tensor_dim0_stride lo32
  g1[6] = 0;                                                        // stride hi | dim1_stride lo
  g1[7] = 0;
  i32x4 z4 = {0, 0, 0, 0};
#if __clang_major__ >= 23
  i32x8 z8 = {0, 0, 0, 0, 0, 0, 0, 0};
  __builtin_amdgcn_tensor_load_to_lds(g0, g1, z4, z4, z8, 0);
#else
  __builtin_amdgcn_tensor_load_to_lds(g0, g1, z4, z4, 0);
#endif
}

// =====================================================================
// Elementwise conversion kernels
// =====================================================================

// W[K][N] fp32 -> Wt[N][K] bf16 (transpose so WMMA B-frag is contiguous in K)
__global__ void k_convT(const float* __restrict__ W, bf16* __restrict__ Wt,
                        int K, int N) {
  size_t i = (size_t)blockIdx.x * blockDim.x + threadIdx.x;
  if (i >= (size_t)K * N) return;
  int k = (int)(i / N);
  int n = (int)(i - (size_t)k * N);
  Wt[(size_t)n * K + k] = (bf16)W[i];
}

// plain fp32 -> bf16
__global__ void k_conv(const float* __restrict__ X, bf16* __restrict__ Y, size_t n) {
  size_t i = (size_t)blockIdx.x * blockDim.x + threadIdx.x;
  if (i < n) Y[i] = (bf16)X[i];
}

// x[b,t,:] = tok_emb[ids[b,t],:] + pos_emb[t,:]
__global__ void k_embed(const int* __restrict__ ids, const float* __restrict__ tok,
                        const float* __restrict__ pos, float* __restrict__ x) {
  size_t i = (size_t)blockIdx.x * blockDim.x + threadIdx.x;
  if (i >= (size_t)NROW * DMODEL) return;
  int bt = (int)(i / DMODEL);
  int d  = (int)(i - (size_t)bt * DMODEL);
  int t  = bt % SEQ;
  x[i] = tok[(size_t)ids[bt] * DMODEL + d] + pos[(size_t)t * DMODEL + d];
}

// split fused qkv fp32 [NROW,3D] into per-head bf16 q/k (row-major) and v^T
__global__ void k_split_qkv(const float* __restrict__ qkv, bf16* __restrict__ qh,
                            bf16* __restrict__ kh, bf16* __restrict__ vth) {
  size_t i = (size_t)blockIdx.x * blockDim.x + threadIdx.x;
  if (i >= (size_t)NROW * DMODEL) return;
  int bt = (int)(i / DMODEL);
  int c  = (int)(i - (size_t)bt * DMODEL);
  int h = c / HDIM, d = c - h * HDIM;
  int b = bt / SEQ, t = bt - b * SEQ;
  size_t base = (size_t)bt * D3;
  size_t rm = ((size_t)(b * NHEAD + h) * SEQ + t) * HDIM + d;           // [b,h,t,d]
  qh[rm] = (bf16)qkv[base + c];
  kh[rm] = (bf16)qkv[base + DMODEL + c];
  vth[((size_t)(b * NHEAD + h) * HDIM + d) * SEQ + t] =                 // [b,h,d,t]
      (bf16)qkv[base + 2 * DMODEL + c];
}

// =====================================================================
// LayerNorm: one 256-thread block per token row, bf16 output
// =====================================================================
__global__ void k_ln(const float* __restrict__ x, const float* __restrict__ w,
                     const float* __restrict__ b, bf16* __restrict__ out) {
  __shared__ float red[256];
  int row = blockIdx.x, tid = threadIdx.x;
  const float* xr = x + (size_t)row * DMODEL;
  float s = 0.f;
  for (int i = tid; i < DMODEL; i += 256) s += xr[i];
  red[tid] = s; __syncthreads();
  for (int st = 128; st; st >>= 1) { if (tid < st) red[tid] += red[tid + st]; __syncthreads(); }
  float mean = red[0] / (float)DMODEL;
  __syncthreads();
  float v = 0.f;
  for (int i = tid; i < DMODEL; i += 256) { float d = xr[i] - mean; v += d * d; }
  red[tid] = v; __syncthreads();
  for (int st = 128; st; st >>= 1) { if (tid < st) red[tid] += red[tid + st]; __syncthreads(); }
  float rs = rsqrtf(red[0] / (float)DMODEL + 1e-5f);
  bf16* o = out + (size_t)row * DMODEL;
  for (int i = tid; i < DMODEL; i += 256)
    o[i] = (bf16)((xr[i] - mean) * rs * w[i] + b[i]);
}

// =====================================================================
// Register-blocked WMMA GEMM with TDM-staged weight tiles.
//   out[M,N] = act( A[M,K](bf16) @ Wt[N,K]^T + bias ) + res
// Workgroup = 8 waves covering a 64x128 C block; each wave owns a 2x2
// arrangement of 16x16 tiles (32x32). B panel (128 rows x 64 K) is DMA'd
// into LDS by the Tensor Data Mover (16 rows per wave), double-barrier
// per K-chunk. A fragments stream from global (L2-resident).
// Requires: M % 64 == 0, K % 64 == 0.
// =====================================================================
#define BLK_M   64
#define BLK_N   128
#define KCHUNK  64
#define LDSPITCH 160   // 64 halves (128B) + 8-DWORD TDM pad = 160B per row

__device__ __forceinline__ v16bf make_afrag(const bf16* arow, int off) {
  v8bf a0 = *(const v8bf*)(arow + off);
  v8bf a1 = *(const v8bf*)(arow + off + 8);
  v16bf a;
#pragma unroll
  for (int j = 0; j < 8; ++j) { a[j] = a0[j]; a[j + 8] = a1[j]; }
  return a;
}

__device__ __forceinline__ void store_tile(v8f c, int m0, int n0, int half, int l15,
                                           const float* bias, const float* res,
                                           float* outF, bf16* outB, int N, int gelu) {
  int n = n0 + l15;
  if (n >= N) return;
  float bv = bias ? bias[n] : 0.0f;
  int mrow = m0 + half * 8;
#pragma unroll
  for (int i = 0; i < 8; ++i) {
    size_t idx = (size_t)(mrow + i) * N + n;
    float v = c[i] + bv;
    if (gelu) v = 0.5f * v * (1.0f + erff(v * 0.70710678118654752f));
    if (res)  v += res[idx];
    if (outF) outF[idx] = v;
    if (outB) outB[idx] = (bf16)v;
  }
}

__global__ void k_gemm(const bf16* __restrict__ A, const bf16* __restrict__ Wt,
                       const float* __restrict__ bias, const float* __restrict__ res,
                       float* __restrict__ outF, bf16* __restrict__ outB,
                       int M, int N, int K, int gelu) {
  __shared__ __align__(32) char smem[BLK_N * LDSPITCH];   // 20 KB
  int nbN  = (N + BLK_N - 1) / BLK_N;
  int bm   = blockIdx.x / nbN;
  int bn   = blockIdx.x - bm * nbN;
  int mBlk = bm * BLK_M;
  int nBlk = bn * BLK_N;
  int wave = threadIdx.x >> 5;          // 0..7
  int lane = threadIdx.x & 31;
  int waveM = wave >> 2;                // 0..1 -> 32-row strip
  int waveN = wave & 3;                 // 0..3 -> 32-col strip
  int half = lane >> 4, l15 = lane & 15;

  int rowsBlk = N - nBlk; if (rowsBlk > BLK_N) rowsBlk = BLK_N;   // valid B rows
  int r0 = wave * 16;                    // this wave's TDM slice of the B panel
  int myrows = rowsBlk - r0; if (myrows > 16) myrows = 16;        // may be <= 0

  const bf16* a0row = A + (size_t)(mBlk + waveM * 32 + l15) * K + half * 16;
  const bf16* a1row = a0row + (size_t)16 * K;
  const char* b0row = smem + (waveN * 32 + l15) * LDSPITCH + half * 32;
  const char* b1row = b0row + 16 * LDSPITCH;
  u32 ldsBase = (u32)(size_t)smem;       // addr[31:0] == LDS byte offset

  v8f c00 = {0.f,0.f,0.f,0.f,0.f,0.f,0.f,0.f};
  v8f c01 = c00, c10 = c00, c11 = c00;

  for (int k0 = 0; k0 < K; k0 += KCHUNK) {
    __syncthreads();                     // previous chunk fully consumed
    if (myrows > 0)                      // wave-uniform; TDM ignores EXEC
      tdm_load_tile(ldsBase + (u32)(r0 * LDSPITCH),
                    Wt + (size_t)(nBlk + r0) * K + k0,
                    KCHUNK, (u32)myrows, (u32)K);
    __builtin_amdgcn_s_wait_tensorcnt(0);
    __syncthreads();                     // B panel visible to all waves
#pragma unroll
    for (int kk = 0; kk < KCHUNK; kk += 32) {
      v16bf af0 = make_afrag(a0row, k0 + kk);
      v16bf af1 = make_afrag(a1row, k0 + kk);
      v16bf bf0 = *(const v16bf*)(b0row + kk * 2);
      v16bf bf1 = *(const v16bf*)(b1row + kk * 2);
      c00 = __builtin_amdgcn_wmma_f32_16x16x32_bf16(false, af0, false, bf0, (short)0, c00, false, false);
      c01 = __builtin_amdgcn_wmma_f32_16x16x32_bf16(false, af0, false, bf1, (short)0, c01, false, false);
      c10 = __builtin_amdgcn_wmma_f32_16x16x32_bf16(false, af1, false, bf0, (short)0, c10, false, false);
      c11 = __builtin_amdgcn_wmma_f32_16x16x32_bf16(false, af1, false, bf1, (short)0, c11, false, false);
    }
  }
  int mW = mBlk + waveM * 32, nW = nBlk + waveN * 32;
  store_tile(c00, mW,      nW,      half, l15, bias, res, outF, outB, N, gelu);
  store_tile(c01, mW,      nW + 16, half, l15, bias, res, outF, outB, N, gelu);
  store_tile(c10, mW + 16, nW,      half, l15, bias, res, outF, outB, N, gelu);
  store_tile(c11, mW + 16, nW + 16, half, l15, bias, res, outF, outB, N, gelu);
}

// =====================================================================
// Attention scores: S[bh][q][k] = (q . k) / 8 via WMMA, skip masked tiles
// =====================================================================
__global__ void k_attn_scores(const bf16* __restrict__ qh, const bf16* __restrict__ kh,
                              float* __restrict__ S) {
  int bh   = blockIdx.y;
  int wave = blockIdx.x * (blockDim.x >> 5) + (threadIdx.x >> 5);
  int lane = threadIdx.x & 31;
  const int tiles = SEQ / 16;                     // 64
  int tm = wave / tiles, tn = wave - tm * tiles;
  if (tm >= tiles) return;
  if (tn > tm) return;                            // entirely above diagonal
  int m0 = tm << 4, n0 = tn << 4;
  int half = lane >> 4, l15 = lane & 15;
  const bf16* arow = qh + (size_t)bh * SEQ * HDIM + (size_t)(m0 + l15) * HDIM + half * 16;
  const bf16* brow = kh + (size_t)bh * SEQ * HDIM + (size_t)(n0 + l15) * HDIM + half * 16;
  v8f c = {0.f,0.f,0.f,0.f,0.f,0.f,0.f,0.f};
  for (int k0 = 0; k0 < HDIM; k0 += 32) {
    v16bf af = make_afrag(arow, k0);
    v16bf bf = *(const v16bf*)(brow + k0);
    c = __builtin_amdgcn_wmma_f32_16x16x32_bf16(false, af, false, bf, (short)0, c, false, false);
  }
  float* Sh = S + (size_t)bh * SEQ * SEQ;
  int n = n0 + l15, mrow = m0 + half * 8;
#pragma unroll
  for (int i = 0; i < 8; ++i)
    Sh[(size_t)(mrow + i) * SEQ + n] = c[i] * 0.125f;   // 1/sqrt(64)
}

// Causal softmax: row = bh*SEQ + q; reads only k<=q, writes full-width bf16 P
__global__ void k_softmax(const float* __restrict__ S, bf16* __restrict__ P) {
  __shared__ float red[256];
  int row = blockIdx.x, tid = threadIdx.x;
  int q = row % SEQ;
  int len = q + 1;
  const float* s = S + (size_t)row * SEQ;
  bf16* p = P + (size_t)row * SEQ;
  float mx = -3.0e38f;
  for (int i = tid; i < len; i += 256) mx = fmaxf(mx, s[i]);
  red[tid] = mx; __syncthreads();
  for (int st = 128; st; st >>= 1) { if (tid < st) red[tid] = fmaxf(red[tid], red[tid + st]); __syncthreads(); }
  mx = red[0]; __syncthreads();
  float sum = 0.f;
  for (int i = tid; i < len; i += 256) sum += expf(s[i] - mx);
  red[tid] = sum; __syncthreads();
  for (int st = 128; st; st >>= 1) { if (tid < st) red[tid] += red[tid + st]; __syncthreads(); }
  float inv = 1.0f / red[0];
  for (int i = tid; i < SEQ; i += 256)
    p[i] = (bf16)((i < len) ? expf(s[i] - mx) * inv : 0.0f);
}

// O = P @ V  (per head), stores fused back into [B,T,D] order as bf16
__global__ void k_attn_out(const bf16* __restrict__ P, const bf16* __restrict__ vth,
                           bf16* __restrict__ oB) {
  int bh   = blockIdx.y;
  int b = bh / NHEAD, h = bh - b * NHEAD;
  int wave = blockIdx.x * (blockDim.x >> 5) + (threadIdx.x >> 5);
  int lane = threadIdx.x & 31;
  const int tilesN = HDIM / 16;                   // 4
  int tm = wave / tilesN, tn = wave - tm * tilesN;
  if (tm >= SEQ / 16) return;
  int m0 = tm << 4, n0 = tn << 4;
  int half = lane >> 4, l15 = lane & 15;
  const bf16* arow = P   + (size_t)bh * SEQ * SEQ  + (size_t)(m0 + l15) * SEQ + half * 16;
  const bf16* brow = vth + (size_t)bh * HDIM * SEQ + (size_t)(n0 + l15) * SEQ + half * 16;
  int kend = (m0 + 16 + 31) & ~31;                // P rows are zero past diagonal
  v8f c = {0.f,0.f,0.f,0.f,0.f,0.f,0.f,0.f};
  for (int k0 = 0; k0 < kend; k0 += 32) {
    v16bf af = make_afrag(arow, k0);
    v16bf bf = *(const v16bf*)(brow + k0);
    c = __builtin_amdgcn_wmma_f32_16x16x32_bf16(false, af, false, bf, (short)0, c, false, false);
  }
  int d = n0 + l15, mrow = m0 + half * 8;
#pragma unroll
  for (int i = 0; i < 8; ++i)
    oB[((size_t)(b * SEQ + mrow + i)) * DMODEL + h * HDIM + d] = (bf16)c[i];
}

// =====================================================================
// Loss: per-row log-sum-exp + NLL, then reduction
// =====================================================================
__global__ void k_row_nll(const float* __restrict__ logits, const int* __restrict__ tgt,
                          float* __restrict__ nll, float* __restrict__ vld) {
  __shared__ float red[256];
  int row = blockIdx.x, tid = threadIdx.x;
  const float* lr = logits + (size_t)row * VOCAB;
  float mx = -3.0e38f;
  for (int i = tid; i < VOCAB; i += 256) mx = fmaxf(mx, lr[i]);
  red[tid] = mx; __syncthreads();
  for (int st = 128; st; st >>= 1) { if (tid < st) red[tid] = fmaxf(red[tid], red[tid + st]); __syncthreads(); }
  mx = red[0]; __syncthreads();
  float s = 0.f;
  for (int i = tid; i < VOCAB; i += 256) s += expf(lr[i] - mx);
  red[tid] = s; __syncthreads();
  for (int st = 128; st; st >>= 1) { if (tid < st) red[tid] += red[tid + st]; __syncthreads(); }
  if (tid == 0) {
    float lse = mx + logf(red[0]);
    int t = tgt[row];
    int ok = (t != -1);
    nll[row] = ok ? (lse - lr[ok ? t : 0]) : 0.0f;
    vld[row] = ok ? 1.0f : 0.0f;
  }
}

__global__ void k_loss(const float* __restrict__ nll, const float* __restrict__ vld,
                       float* __restrict__ out) {
  __shared__ float rn[256], rv[256];
  int tid = threadIdx.x;
  float sn = 0.f, sv = 0.f;
  for (int i = tid; i < NROW; i += 256) { sn += nll[i]; sv += vld[i]; }
  rn[tid] = sn; rv[tid] = sv; __syncthreads();
  for (int st = 128; st; st >>= 1) {
    if (tid < st) { rn[tid] += rn[tid + st]; rv[tid] += rv[tid + st]; }
    __syncthreads();
  }
  if (tid == 0) out[0] = rn[0] / fmaxf(rv[0], 1.0f);
}

// =====================================================================
// Host-side orchestration
// =====================================================================
static inline void launch_gemm(const bf16* A, const bf16* Wt, const float* bias,
                               const float* res, float* outF, bf16* outB,
                               int M, int N, int K, int gelu, hipStream_t s) {
  int nbM = M / BLK_M;
  int nbN = CEILDIV(N, BLK_N);
  k_gemm<<<nbM * nbN, 256, 0, s>>>(A, Wt, bias, res, outF, outB, M, N, K, gelu);
}

extern "C" void kernel_launch(void* const* d_in, const int* in_sizes, int n_in,
                              void* d_out, int out_size, void* d_ws, size_t ws_size,
                              hipStream_t stream) {
  const int*   ids  = (const int*)d_in[0];
  const int*   tgt  = (const int*)d_in[1];
  const float* tok  = (const float*)d_in[2];
  const float* pos  = (const float*)d_in[3];
  const float* ln1w = (const float*)d_in[4];
  const float* ln1b = (const float*)d_in[5];
  const float* qkvw = (const float*)d_in[6];
  const float* qkvb = (const float*)d_in[7];
  const float* apw  = (const float*)d_in[8];
  const float* apb  = (const float*)d_in[9];
  const float* ln2w = (const float*)d_in[10];
  const float* ln2b = (const float*)d_in[11];
  const float* fcw  = (const float*)d_in[12];
  const float* fcb  = (const float*)d_in[13];
  const float* prw  = (const float*)d_in[14];
  const float* prb  = (const float*)d_in[15];
  const float* lnfw = (const float*)d_in[16];
  const float* lnfb = (const float*)d_in[17];

  float* lossp  = (float*)d_out;
  float* logits = lossp + 1;                      // [NROW, VOCAB]

  // ---- carve workspace (256B aligned) ----
  char* p = (char*)d_ws;
  auto carve = [&](size_t bytes) -> char* {
    char* r = p; p += (bytes + 255) & ~(size_t)255; return r;
  };
  bf16* wqkvT = (bf16*)carve((size_t)NLAYER * D3 * DMODEL * sizeof(bf16));
  bf16* wattT = (bf16*)carve((size_t)NLAYER * DMODEL * DMODEL * sizeof(bf16));
  bf16* wfcT  = (bf16*)carve((size_t)NLAYER * D4 * DMODEL * sizeof(bf16));
  bf16* wmlT  = (bf16*)carve((size_t)NLAYER * DMODEL * D4 * sizeof(bf16));
  bf16* embB  = (bf16*)carve((size_t)VOCAB * DMODEL * sizeof(bf16));
  float* x    = (float*)carve((size_t)NROW * DMODEL * sizeof(float));
  bf16* hB    = (bf16*)carve((size_t)NROW * DMODEL * sizeof(bf16));
  float* qkvF = (float*)carve((size_t)NROW * D3 * sizeof(float));
  bf16* qh    = (bf16*)carve((size_t)NROW * DMODEL * sizeof(bf16));
  bf16* kh    = (bf16*)carve((size_t)NROW * DMODEL * sizeof(bf16));
  bf16* vth   = (bf16*)carve((size_t)NROW * DMODEL * sizeof(bf16));
  float* S    = (float*)carve((size_t)BATCH * NHEAD * SEQ * SEQ * sizeof(float));
  bf16* P     = (bf16*)carve((size_t)BATCH * NHEAD * SEQ * SEQ * sizeof(bf16));
  bf16* oB    = (bf16*)carve((size_t)NROW * DMODEL * sizeof(bf16));
  bf16* actB  = (bf16*)carve((size_t)NROW * D4 * sizeof(bf16));
  bf16* xfB   = (bf16*)carve((size_t)NROW * DMODEL * sizeof(bf16));
  float* nllA = (float*)carve((size_t)NROW * sizeof(float));
  float* vldA = (float*)carve((size_t)NROW * sizeof(float));
  (void)ws_size; (void)in_sizes; (void)n_in; (void)out_size;

  // ---- weight conversion (fp32 -> transposed bf16) ----
  for (int l = 0; l < NLAYER; ++l) {
    {
      size_t n = (size_t)DMODEL * D3;
      k_convT<<<(int)CEILDIV(n, 256), 256, 0, stream>>>(
          qkvw + (size_t)l * n, wqkvT + (size_t)l * n, DMODEL, D3);
    }
    {
      size_t n = (size_t)DMODEL * DMODEL;
      k_convT<<<(int)CEILDIV(n, 256), 256, 0, stream>>>(
          apw + (size_t)l * n, wattT + (size_t)l * n, DMODEL, DMODEL);
    }
    {
      size_t n = (size_t)DMODEL * D4;
      k_convT<<<(int)CEILDIV(n, 256), 256, 0, stream>>>(
          fcw + (size_t)l * n, wfcT + (size_t)l * n, DMODEL, D4);
    }
    {
      size_t n = (size_t)D4 * DMODEL;
      k_convT<<<(int)CEILDIV(n, 256), 256, 0, stream>>>(
          prw + (size_t)l * n, wmlT + (size_t)l * n, D4, DMODEL);
    }
  }
  {
    size_t n = (size_t)VOCAB * DMODEL;
    k_conv<<<(int)CEILDIV(n, 256), 256, 0, stream>>>(tok, embB, n);
  }

  // ---- embedding ----
  k_embed<<<(int)CEILDIV((size_t)NROW * DMODEL, 256), 256, 0, stream>>>(ids, tok, pos, x);

  // ---- transformer blocks ----
  for (int l = 0; l < NLAYER; ++l) {
    k_ln<<<NROW, 256, 0, stream>>>(x, ln1w + l * DMODEL, ln1b + l * DMODEL, hB);
    launch_gemm(hB, wqkvT + (size_t)l * D3 * DMODEL, qkvb + (size_t)l * D3,
                nullptr, qkvF, nullptr, NROW, D3, DMODEL, 0, stream);
    k_split_qkv<<<(int)CEILDIV((size_t)NROW * DMODEL, 256), 256, 0, stream>>>(qkvF, qh, kh, vth);
    k_attn_scores<<<dim3(512, BATCH * NHEAD), 256, 0, stream>>>(qh, kh, S);
    k_softmax<<<BATCH * NHEAD * SEQ, 256, 0, stream>>>(S, P);
    k_attn_out<<<dim3(32, BATCH * NHEAD), 256, 0, stream>>>(P, vth, oB);
    launch_gemm(oB, wattT + (size_t)l * DMODEL * DMODEL, apb + (size_t)l * DMODEL,
                x, x, nullptr, NROW, DMODEL, DMODEL, 0, stream);
    k_ln<<<NROW, 256, 0, stream>>>(x, ln2w + l * DMODEL, ln2b + l * DMODEL, hB);
    launch_gemm(hB, wfcT + (size_t)l * D4 * DMODEL, fcb + (size_t)l * D4,
                nullptr, nullptr, actB, NROW, D4, DMODEL, 1, stream);
    launch_gemm(actB, wmlT + (size_t)l * DMODEL * D4, prb + (size_t)l * DMODEL,
                x, x, nullptr, NROW, DMODEL, D4, 0, stream);
  }

  // ---- final LN + tied lm-head -> logits straight into d_out ----
  k_ln<<<NROW, 256, 0, stream>>>(x, lnfw, lnfb, xfB);
  launch_gemm(xfB, embB, nullptr, nullptr, logits, nullptr, NROW, VOCAB, DMODEL, 0, stream);

  // ---- loss ----
  k_row_nll<<<NROW, 256, 0, stream>>>(logits, tgt, nllA, vldA);
  k_loss<<<1, 256, 0, stream>>>(nllA, vldA, lossp);
}
